// HierarchicalAutoEncoderLayer_26319559590069
// MI455X (gfx1250) — matrix operand order repaired and checked by reading, not compiled, not run on hardware
//
#include <hip/hip_runtime.h>

// ---------------- problem constants (from reference) ----------------
#define BATCH  4096
#define N_SAE  16
#define D_DATA 768
#define D_DICT 256
#define MTILE  32          // batch rows per fused workgroup
#define YSTRIDE 1536       // staging rows per sae (expected ~1024 active, +18 sigma margin)

// ---------------- types ----------------
typedef __attribute__((ext_vector_type(16))) __bf16 v16bf;
typedef __attribute__((ext_vector_type(8)))  __bf16 v8bf;
typedef __attribute__((ext_vector_type(8)))  float  v8f;

union V16 {
    v16bf v;
    v8bf  h[2];
};

__device__ __forceinline__ unsigned short f2bf(float f) {
    // round-to-nearest-even fp32 -> bf16 (bit pattern)
    unsigned u = __builtin_bit_cast(unsigned, f);
    unsigned r = u + 0x7FFFu + ((u >> 16) & 1u);
    return (unsigned short)(r >> 16);
}

__device__ __forceinline__ float bf2f(unsigned short h) {
    unsigned u = ((unsigned)h) << 16;
    return __builtin_bit_cast(float, u);
}

// ---------------- prep kernels ----------------
__global__ void sae_cvt_x(const float4* __restrict__ x, unsigned short* __restrict__ xbf, int n4) {
    int i = blockIdx.x * blockDim.x + threadIdx.x;
    if (i >= n4) return;
    float4 v = x[i];
    ushort4 w;
    w.x = f2bf(v.x); w.y = f2bf(v.y); w.z = f2bf(v.z); w.w = f2bf(v.w);
    *(ushort4*)(xbf + i * 4) = w;
}

// Tiled transpose+convert: in[s][R][C] f32 -> out[s][C][R] bf16.
// Coalesced f32 reads, coalesced bf16 writes via a 32x33 LDS tile.
__global__ __launch_bounds__(256) void sae_transpose_cvt(const float* __restrict__ in,
                                                         unsigned short* __restrict__ outp,
                                                         int R, int C) {
    __shared__ float tile[32][33];
    const int s  = blockIdx.z;
    const int r0 = blockIdx.x * 32;
    const int c0 = blockIdx.y * 32;
    const int tx = threadIdx.x & 31;
    const int ty = threadIdx.x >> 5;   // 0..7
    const float*    src = in  + (size_t)s * R * C;
    unsigned short* dst = outp + (size_t)s * R * C;
#pragma unroll
    for (int i = 0; i < 4; ++i) {
        tile[ty + i * 8][tx] = src[(size_t)(r0 + ty + i * 8) * C + c0 + tx];
    }
    __syncthreads();
#pragma unroll
    for (int i = 0; i < 4; ++i) {
        dst[(size_t)(c0 + ty + i * 8) * R + r0 + tx] = f2bf(tile[tx][ty + i * 8]);
    }
}

// cbias[s][e] = sum_k b_dec[s][k] * W_enc[s][k][e]   (reads coalesced over e)
__global__ void sae_enc_bias(const float* __restrict__ W_enc, const float* __restrict__ b_dec,
                             float* __restrict__ cbias) {
    int s = blockIdx.x;
    int e = threadIdx.x;
    const float* w  = W_enc + (size_t)s * D_DATA * D_DICT;
    const float* bd = b_dec + (size_t)s * D_DATA;
    float acc = 0.f;
    for (int k = 0; k < D_DATA; ++k) acc += bd[k] * w[(size_t)k * D_DICT + e];
    cbias[s * D_DICT + e] = acc;
}

// Deterministic per-sae compaction of active rows (gate != 0) + inverse map posIdx[b][s].
__global__ void sae_build_index(const float* __restrict__ gate, int* __restrict__ counts,
                                int* __restrict__ idx, int* __restrict__ posIdx) {
    int s    = blockIdx.x;
    int lane = threadIdx.x;          // wave32
    int base = 0;
    for (int c = 0; c < BATCH; c += 32) {
        int b    = c + lane;
        float g  = gate[b * N_SAE + s];
        bool  p  = (g != 0.0f);
        unsigned mask = (unsigned)__ballot(p);
        int off = __popc(mask & ((1u << lane) - 1u));
        if (p) idx[s * BATCH + base + off] = b;
        posIdx[b * N_SAE + s] = p ? (base + off) : -1;
        base += __popc(mask);
    }
    if (lane == 0) counts[s] = base;
}

// ---------------- fused encoder->relu->gate->decoder, per (sae, 32-row tile) ----------------
__global__ __launch_bounds__(512) void sae_fused(
    const unsigned short* __restrict__ xbf,     // [BATCH, D_DATA] bf16
    const unsigned short* __restrict__ wencT,   // [S, E, K] bf16
    const unsigned short* __restrict__ wdecT,   // [S, D, E] bf16
    const float* __restrict__ gate,             // [BATCH, S]
    const float* __restrict__ b_enc,            // [S, E]
    const float* __restrict__ cbias,            // [S, E]
    const int*   __restrict__ counts,           // [S]
    const int*   __restrict__ idx,              // [S, BATCH]
    unsigned short* __restrict__ ypair)         // [S, YSTRIDE, D_DATA] bf16 staging
{
    __shared__ unsigned short sX[MTILE * D_DATA];      // 48 KB: gathered x rows (bf16)
    __shared__ unsigned short sActs[MTILE * D_DICT];   // 16 KB: gated activations (bf16)
    __shared__ int            sRowB[MTILE];
    __shared__ int            sRowValid[MTILE];
    __shared__ float          sGate[MTILE];

    const int s    = blockIdx.y;
    const int M    = counts[s];
    const int row0 = blockIdx.x * MTILE;
    if (row0 >= M) return;                             // uniform across block

    const int tid = threadIdx.x;
    if (tid < MTILE) {
        int r  = row0 + tid;
        int rc = (r < M) ? r : (M - 1);
        int b  = idx[s * BATCH + rc];
        sRowB[tid]     = b;
        sRowValid[tid] = (r < M);
        sGate[tid]     = (r < M) ? gate[b * N_SAE + s] : 0.0f;
    }
    __syncthreads();

    // stage gathered x rows into LDS (16B chunks; 32 rows * 96 chunks = 3072, 6 per thread)
#pragma unroll
    for (int i = 0; i < 6; ++i) {
        int c   = tid + i * 512;
        int row = c / 96;
        int off = (c % 96) * 8;
        const uint4* src = (const uint4*)(xbf + (size_t)sRowB[row] * D_DATA + off);
        *(uint4*)(&sX[row * D_DATA + off]) = *src;
    }
    __syncthreads();

    const int wave = tid >> 5;    // 0..15
    const int lane = tid & 31;
    const int lo   = lane & 15;
    const int hi   = lane >> 4;

    // ---------------- encoder: [32 x 768] @ [768 x 256] ----------------
    v8f accE[2] = {};
    const unsigned short* bG = wencT + ((size_t)s * D_DICT + wave * 16 + lo) * D_DATA + 16 * hi;

    for (int k = 0; k < D_DATA; k += 32) {
        __builtin_prefetch(bG + k + 256, 0, 1);       // global_prefetch of upcoming B slab
        V16 B;
        B.v = *(const v16bf*)(bG + k);
#pragma unroll
        for (int sub = 0; sub < 2; ++sub) {
            const unsigned short* ar = &sX[(sub * 16 + lo) * D_DATA];
            V16 A;
            A.h[0] = *(const v8bf*)(ar + k + 16 * hi);
            A.h[1] = *(const v8bf*)(ar + k + 16 * hi + 8);
            accE[sub] = __builtin_amdgcn_wmma_f32_16x16x32_bf16(false, A.v, false, B.v,
                                                                (short)0, accE[sub], false, false);
        }
    }

    {
        const int   col  = wave * 16 + lo;
        const float bias = b_enc[s * D_DICT + col] - cbias[s * D_DICT + col];
#pragma unroll
        for (int sub = 0; sub < 2; ++sub) {
#pragma unroll
            for (int v = 0; v < 8; ++v) {
                const int row = sub * 16 + v + 8 * hi;
                float a = accE[sub][v] + bias;
                a = a > 0.0f ? a : 0.0f;
                sActs[row * D_DICT + col] = f2bf(a * sGate[row]);
            }
        }
    }
    __syncthreads();

    // ---------------- decoder: [32 x 256] @ [256 x 768] ----------------
    v8f accD[2][3] = {};
    for (int k = 0; k < D_DICT; k += 32) {
        V16 A[2];
#pragma unroll
        for (int sub = 0; sub < 2; ++sub) {
            const unsigned short* ar = &sActs[(sub * 16 + lo) * D_DICT];
            A[sub].h[0] = *(const v8bf*)(ar + k + 16 * hi);
            A[sub].h[1] = *(const v8bf*)(ar + k + 16 * hi + 8);
        }
#pragma unroll
        for (int t = 0; t < 3; ++t) {
            const int col = wave * 48 + t * 16 + lo;
            V16 B;
            B.v = *(const v16bf*)(wdecT + ((size_t)s * D_DATA + col) * D_DICT + 16 * hi + k);
#pragma unroll
            for (int sub = 0; sub < 2; ++sub) {
                accD[sub][t] = __builtin_amdgcn_wmma_f32_16x16x32_bf16(false, A[sub].v, false, B.v,
                                                                       (short)0, accD[sub][t],
                                                                       false, false);
            }
        }
    }

    // store decoded rows (bf16) into per-sae staging; no atomics
#pragma unroll
    for (int t = 0; t < 3; ++t) {
        const int col = wave * 48 + t * 16 + lo;
#pragma unroll
        for (int sub = 0; sub < 2; ++sub) {
#pragma unroll
            for (int v = 0; v < 8; ++v) {
                const int row = sub * 16 + v + 8 * hi;
                if (sRowValid[row]) {
                    const int r = row0 + row;
                    ypair[((size_t)s * YSTRIDE + r) * D_DATA + col] = f2bf(accD[sub][t][v]);
                }
            }
        }
    }
}

// ---------------- deterministic per-batch-row reduction ----------------
__global__ __launch_bounds__(256) void sae_reduce(
    const unsigned short* __restrict__ ypair,   // [S, YSTRIDE, D_DATA] bf16
    const int*   __restrict__ posIdx,           // [BATCH, S]
    const float* __restrict__ b_dec,            // [S, D_DATA]
    float*       __restrict__ out)              // [BATCH, D_DATA]
{
    const int b = blockIdx.x;
    const int t = threadIdx.x;                  // 0..255, covers 3 columns each
    float a0 = 0.f, a1 = 0.f, a2 = 0.f;
#pragma unroll
    for (int s = 0; s < N_SAE; ++s) {
        const int pos = posIdx[b * N_SAE + s];
        if (pos >= 0) {
            const unsigned short* yr = ypair + ((size_t)s * YSTRIDE + pos) * D_DATA;
            const float*          bd = b_dec + s * D_DATA;
            a0 += bf2f(yr[t])       + bd[t];
            a1 += bf2f(yr[t + 256]) + bd[t + 256];
            a2 += bf2f(yr[t + 512]) + bd[t + 512];
        }
    }
    float* o = out + (size_t)b * D_DATA;
    o[t] = a0; o[t + 256] = a1; o[t + 512] = a2;
}

// ---------------- host launcher ----------------
extern "C" void kernel_launch(void* const* d_in, const int* in_sizes, int n_in,
                              void* d_out, int out_size, void* d_ws, size_t ws_size,
                              hipStream_t stream) {
    (void)in_sizes; (void)n_in; (void)out_size; (void)ws_size;

    const float* x     = (const float*)d_in[0];
    const float* gate  = (const float*)d_in[1];
    const float* W_enc = (const float*)d_in[2];
    const float* b_enc = (const float*)d_in[3];
    const float* W_dec = (const float*)d_in[4];
    const float* b_dec = (const float*)d_in[5];
    float* out = (float*)d_out;

    // workspace carve-up (all 256B aligned)
    char* ws = (char*)d_ws;
    size_t off = 0;
    int* counts = (int*)(ws + off);                      off += 256;
    int* idx    = (int*)(ws + off);                      off += (size_t)N_SAE * BATCH * sizeof(int);
    int* posIdx = (int*)(ws + off);                      off += (size_t)BATCH * N_SAE * sizeof(int);
    unsigned short* xbf   = (unsigned short*)(ws + off); off += (size_t)BATCH * D_DATA * 2;
    unsigned short* wencT = (unsigned short*)(ws + off); off += (size_t)N_SAE * D_DATA * D_DICT * 2;
    unsigned short* wdecT = (unsigned short*)(ws + off); off += (size_t)N_SAE * D_DATA * D_DICT * 2;
    float* cbias = (float*)(ws + off);                   off += (size_t)N_SAE * D_DICT * sizeof(float);
    unsigned short* ypair = (unsigned short*)(ws + off); off += (size_t)N_SAE * YSTRIDE * D_DATA * 2;

    const int nX4 = BATCH * D_DATA / 4;                  // 786432

    sae_cvt_x<<<(nX4 + 255) / 256, 256, 0, stream>>>((const float4*)x, xbf, nX4);

    // W_enc [S,768,256] -> [S,256,768]
    sae_transpose_cvt<<<dim3(D_DATA / 32, D_DICT / 32, N_SAE), 256, 0, stream>>>(W_enc, wencT,
                                                                                 D_DATA, D_DICT);
    // W_dec [S,256,768] -> [S,768,256]
    sae_transpose_cvt<<<dim3(D_DICT / 32, D_DATA / 32, N_SAE), 256, 0, stream>>>(W_dec, wdecT,
                                                                                 D_DICT, D_DATA);

    sae_enc_bias<<<N_SAE, D_DICT, 0, stream>>>(W_enc, b_dec, cbias);
    sae_build_index<<<N_SAE, 32, 0, stream>>>(gate, counts, idx, posIdx);

    dim3 grid(BATCH / MTILE, N_SAE);   // (128, 16); tiles past active count early-exit
    sae_fused<<<grid, 512, 0, stream>>>(xbf, wencT, wdecT, gate, b_enc, cbias,
                                        counts, idx, ypair);

    sae_reduce<<<BATCH, 256, 0, stream>>>(ypair, posIdx, b_dec, out);
}
